// BasicBlock_multi_60258391163148
// MI455X (gfx1250) — matrix operand order repaired and checked by reading, not compiled, not run on hardware
//
#include <hip/hip_runtime.h>
#include <cstdint>
#include <cstddef>

// ---------- types ----------
typedef __bf16 bf16_t;
typedef __attribute__((ext_vector_type(16))) __bf16 v16bf;
typedef __attribute__((ext_vector_type(8)))  float  v8f;
typedef unsigned int u32x4 __attribute__((ext_vector_type(4)));
typedef int v4i __attribute__((ext_vector_type(4)));

// ---------- problem constants ----------
#define C_CH   128
#define H_DIM  56
#define W_DIM  56
#define HW     (H_DIM * W_DIM)          // 3136
#define N_BATCH 64
#define NHW    (N_BATCH * HW)           // 200704
#define M_ROWS 32                        // spatial rows per block (2 WMMA M-tiles)
#define LDS_ROW 272                      // 256B bf16 data + 16B pad (16B aligned)
#define WPK_ELEMS (9 * 4 * 8 * 32 * 16)  // 147456 bf16 per weight tensor

// Async global->LDS (gfx1250 GLOBAL_LOAD_ASYNC_TO_LDS_B128, ASYNCcnt) if the
// toolchain exposes the builtin; otherwise fall back to load+ds_store.
#if defined(__HIP_DEVICE_COMPILE__) && __has_builtin(__builtin_amdgcn_global_load_async_to_lds_b128)
#define USE_ASYNC_LDS 1
#else
#define USE_ASYNC_LDS 0
#endif

// =====================================================================
// Prep kernel 1: pack conv weights [O=128][I=128][3][3] f32 into exact
// per-lane WMMA B-fragment order (bf16), layout [tap][kchunk][coblk][lane][16].
// Dense 16-bit B (32x16, KxN): lane L -> col N = L%16; this packing (not the ISA table order) puts
// K=0..15 in lanes 0-15, K=16..31 in lanes 16-31 (2 K-values per VGPR).
// =====================================================================
__global__ void pack_weights_kernel(const float* __restrict__ w1,
                                    const float* __restrict__ w2,
                                    bf16_t* __restrict__ p1,
                                    bf16_t* __restrict__ p2) {
    int p = blockIdx.x * blockDim.x + threadIdx.x;
    if (p >= WPK_ELEMS) return;
    int j    = p & 15;          // element within lane's 32B fragment
    int lane = (p >> 4) & 31;
    int bco  = (p >> 9) & 7;    // co block (8 x 16 = 128)
    int c    = (p >> 12) & 3;   // k chunk of 32 within a tap
    int t    = p >> 14;         // tap 0..8
    int co = bco * 16 + (lane & 15);
    int k  = 32 * c + ((lane & 16) ? 16 : 0) + j;   // ci within the tap
    size_t s = ((size_t)co * C_CH + (size_t)k) * 9 + (size_t)t; // [O][I][3][3]
    p1[p] = (bf16_t)w1[s];
    p2[p] = (bf16_t)w2[s];
}

// =====================================================================
// Prep kernel 2: x NCHW f32 -> NHWC bf16 (channel-contiguous K chunks)
// =====================================================================
__global__ void x_to_nhwc_bf16_kernel(const float* __restrict__ x,
                                      bf16_t* __restrict__ xbf) {
    size_t i = (size_t)blockIdx.x * blockDim.x + threadIdx.x;
    if (i >= (size_t)NHW * C_CH) return;
    int    ci = (int)(i % C_CH);
    size_t s  = (i / C_CH) % HW;
    size_t n  = i / ((size_t)C_CH * HW);
    xbf[i] = (bf16_t)x[(n * C_CH + (size_t)ci) * HW + s];
}

// =====================================================================
// Prep kernel 3: fold BN into per-channel scale/shift
// prm layout: [scale1 | shift1 | scale2 | shift2], 128 floats each
// =====================================================================
__global__ void bn_prep_kernel(const float* g1, const float* b1,
                               const float* m1, const float* v1,
                               const float* g2, const float* b2,
                               const float* m2, const float* v2,
                               float* __restrict__ prm) {
    int c = threadIdx.x;
    if (c < C_CH) {
        float s1 = g1[c] * rsqrtf(v1[c] + 1e-5f);
        prm[c]             = s1;
        prm[C_CH + c]      = b1[c] - m1[c] * s1;
        float s2 = g2[c] * rsqrtf(v2[c] + 1e-5f);
        prm[2 * C_CH + c]  = s2;
        prm[3 * C_CH + c]  = b2[c] - m2[c] * s2;
    }
}

// =====================================================================
// Fused implicit-GEMM conv3x3 + BN (+ residual) + ReLU, bf16 WMMA.
// Block: 256 threads = 8 waves. Block owns a 32-row spatial tile (M) and all
// 128 output channels (wave w -> co block w). Each wave holds TWO 16x16 f32
// accumulators (M halves) and reuses each B fragment across both -> 2x
// arithmetic intensity on weights. A tiles (32x128 bf16 per tap) are staged
// in LDS via async global->LDS when available.
// =====================================================================
template <int SECOND>
__global__ __launch_bounds__(256)
void conv_wmma_kernel(const bf16_t* __restrict__ src,    // NHWC bf16 activations
                      const bf16_t* __restrict__ wpk,    // packed B fragments
                      const float*  __restrict__ scale,  // per-co BN scale
                      const float*  __restrict__ shift,  // per-co BN shift
                      const float*  __restrict__ xres,   // NCHW f32 residual (SECOND)
                      bf16_t* __restrict__ ybf,          // NHWC bf16 out (!SECOND)
                      float*  __restrict__ yf)           // NCHW f32 out (SECOND)
{
    __shared__ __align__(16) unsigned char sA[M_ROWS * LDS_ROW];

    const int tid  = (int)threadIdx.x;
    const int wave = tid >> 5;
    const int lane = tid & 31;
    const int mBase = (int)blockIdx.x * M_ROWS;

    // Staging role: each thread stages 2 rows x one 16B segment per tap.
    const int srow0 = tid >> 4;          // 0..15
    const int srow1 = srow0 + 16;        // 16..31
    const int sseg  = tid & 15;          // 16B segment of the 256B row
    int sn[2], shh[2], sww[2];
    {
        const int p0 = mBase + srow0;
        sn[0] = p0 / HW; const int r0 = p0 % HW; shh[0] = r0 / W_DIM; sww[0] = r0 % W_DIM;
        const int p1 = mBase + srow1;
        sn[1] = p1 / HW; const int r1 = p1 % HW; shh[1] = r1 / W_DIM; sww[1] = r1 % W_DIM;
    }

    const int laneHalf = lane >> 4;     // 0 or 1
    const int col      = lane & 15;

    v8f acc0 = {};                      // M rows mBase+0..15
    v8f acc1 = {};                      // M rows mBase+16..31

    for (int t = 0; t < 9; ++t) {
        const int dy = t / 3 - 1;
        const int dx = t % 3 - 1;

        // ---- stage A tile: 32 rows x 128 ci bf16 (zero-fill at halo) ----
        #pragma unroll
        for (int rr = 0; rr < 2; ++rr) {
            const int srow = rr ? srow1 : srow0;
            const int hs = shh[rr] + dy;
            const int ws = sww[rr] + dx;
            const bool valid = ((unsigned)hs < (unsigned)H_DIM) &&
                               ((unsigned)ws < (unsigned)W_DIM);
            u32x4* lptr = reinterpret_cast<u32x4*>(&sA[srow * LDS_ROW + sseg * 16]);
            const u32x4* gp = reinterpret_cast<const u32x4*>(
                src + (((size_t)sn[rr] * HW) + (size_t)hs * W_DIM + ws) * C_CH + sseg * 8);
#if USE_ASYNC_LDS
            if (valid) {
                __builtin_amdgcn_global_load_async_to_lds_b128(
                    (v4i*)gp, (v4i*)lptr, 0, 0);
            } else {
                *lptr = (u32x4)0;        // ds_store_b128 zero fill
            }
#else
            u32x4 v = (u32x4)0;
            if (valid) v = *gp;          // global_load_b128
            *lptr = v;                   // ds_store_b128
#endif
        }
#if USE_ASYNC_LDS
#if __has_builtin(__builtin_amdgcn_s_wait_asynccnt)
        __builtin_amdgcn_s_wait_asynccnt(0);
#else
        asm volatile("s_wait_asynccnt 0x0" ::: "memory");
#endif
#endif
        __syncthreads();

        // prefetch next tap's packed weights toward L2
        if (t < 8) {
            __builtin_prefetch(wpk + (((size_t)(t + 1) * 4) * 8 + wave) * 32 * 16, 0, 0);
        }

        // ---- 4 K-chunks of 32 per tap; B fragment reused for both M halves ----
        #pragma unroll
        for (int c = 0; c < 4; ++c) {
            union { u32x4 q[2]; v16bf v; } af0, af1, bfr;
            // B fragment: pre-packed, lane-contiguous 32B (2x b128, coalesced)
            const u32x4* wp = reinterpret_cast<const u32x4*>(
                wpk + ((((size_t)t * 4 + c) * 8 + wave) * 32 + lane) * 16);
            bfr.q[0] = wp[0];
            bfr.q[1] = wp[1];
            // A fragments, K order matching packed B (contiguous):
            // lane<16: K 0..15 @ +0/+16 ; lane>=16: K16..31 @ +32/+48
            const int ab0 = col * LDS_ROW + c * 64 + laneHalf * 32;
            const int ab1 = (col + 16) * LDS_ROW + c * 64 + laneHalf * 32;
            af0.q[0] = *reinterpret_cast<const u32x4*>(&sA[ab0]);
            af0.q[1] = *reinterpret_cast<const u32x4*>(&sA[ab0 + 16]);
            af1.q[0] = *reinterpret_cast<const u32x4*>(&sA[ab1]);
            af1.q[1] = *reinterpret_cast<const u32x4*>(&sA[ab1 + 16]);
            acc0 = __builtin_amdgcn_wmma_f32_16x16x32_bf16(
                false, af0.v, false, bfr.v, (short)0, acc0, false, false);
            acc1 = __builtin_amdgcn_wmma_f32_16x16x32_bf16(
                false, af1.v, false, bfr.v, (short)0, acc1, false, false);
        }
        __syncthreads();   // protect sA before next tap overwrites
    }

    // ---- epilogue: BN (+residual) + ReLU ----
    const int co = wave * 16 + col;
    const float sc = scale[co];
    const float sh = shift[co];

    #pragma unroll
    for (int half = 0; half < 2; ++half) {
        const v8f& acc = half ? acc1 : acc0;
        #pragma unroll
        for (int r = 0; r < 8; ++r) {
            const int m   = half * 16 + r + laneHalf * 8;  // C/D: VGPR r -> M=r / r+8
            const int pos = mBase + m;
            float val = acc[r] * sc + sh;
            if (SECOND) {
                const int n = pos / HW;
                const int s = pos % HW;
                const size_t oidx = ((size_t)n * C_CH + co) * HW + s;   // NCHW
                val += xres[oidx];
                val = fmaxf(val, 0.0f);
                yf[oidx] = val;
            } else {
                val = fmaxf(val, 0.0f);
                ybf[(size_t)pos * C_CH + co] = (bf16_t)val;             // NHWC bf16
            }
        }
    }
}

// =====================================================================
// Host launcher
// =====================================================================
extern "C" void kernel_launch(void* const* d_in, const int* in_sizes, int n_in,
                              void* d_out, int out_size, void* d_ws, size_t ws_size,
                              hipStream_t stream) {
    const float* x  = (const float*)d_in[0];
    const float* w1 = (const float*)d_in[1];
    const float* g1 = (const float*)d_in[2];
    const float* b1 = (const float*)d_in[3];
    const float* m1 = (const float*)d_in[4];
    const float* v1 = (const float*)d_in[5];
    const float* w2 = (const float*)d_in[6];
    const float* g2 = (const float*)d_in[7];
    const float* b2 = (const float*)d_in[8];
    const float* m2 = (const float*)d_in[9];
    const float* v2 = (const float*)d_in[10];
    float* out = (float*)d_out;

    // ---- workspace carve-up (all offsets 256B-aligned) ----
    uint8_t* ws = (uint8_t*)d_ws;
    size_t off = 0;
    const size_t ACT_BYTES = (size_t)NHW * C_CH * sizeof(bf16_t);   // 51,380,224
    bf16_t* xbf = (bf16_t*)(ws + off); off += ACT_BYTES;
    bf16_t* y1  = (bf16_t*)(ws + off); off += ACT_BYTES;
    bf16_t* p1  = (bf16_t*)(ws + off); off += (size_t)WPK_ELEMS * sizeof(bf16_t);
    bf16_t* p2  = (bf16_t*)(ws + off); off += (size_t)WPK_ELEMS * sizeof(bf16_t);
    float*  prm = (float*)(ws + off);  off += 4 * C_CH * sizeof(float);
    (void)ws_size; (void)in_sizes; (void)n_in; (void)out_size;

    // ---- prep ----
    pack_weights_kernel<<<(WPK_ELEMS + 255) / 256, 256, 0, stream>>>(w1, w2, p1, p2);

    const size_t nElems = (size_t)NHW * C_CH;
    x_to_nhwc_bf16_kernel<<<(unsigned)((nElems + 255) / 256), 256, 0, stream>>>(x, xbf);

    bn_prep_kernel<<<1, 128, 0, stream>>>(g1, b1, m1, v1, g2, b2, m2, v2, prm);

    // ---- fused convs ----
    const unsigned grid = NHW / M_ROWS;   // 6272
    conv_wmma_kernel<0><<<grid, 256, 0, stream>>>(
        xbf, p1, prm, prm + C_CH, nullptr, y1, nullptr);
    conv_wmma_kernel<1><<<grid, 256, 0, stream>>>(
        y1, p2, prm + 2 * C_CH, prm + 3 * C_CH, x, nullptr, out);
}
